// GroupedQueryAttention_18975165514194
// MI455X (gfx1250) — matrix operand order repaired and checked by reading, not compiled, not run on hardware
//
#include <hip/hip_runtime.h>
#include <hip/hip_bf16.h>
#include <stdint.h>

// ---------------------------------------------------------------------------
// CDNA5 (gfx1250) wave32 WMMA bf16 building blocks
// ---------------------------------------------------------------------------
typedef __attribute__((ext_vector_type(16))) __bf16 v16bf;
typedef __attribute__((ext_vector_type(8)))  float  v8f;
typedef unsigned int v4u __attribute__((ext_vector_type(4)));
typedef int          v8i __attribute__((ext_vector_type(8)));
typedef int          v4i __attribute__((ext_vector_type(4)));

union FragU {
  v16bf v;
  uint4 q[2];
  unsigned short u[16];
};

static __device__ __forceinline__ unsigned short f32_to_bf16_rne(float f) {
  unsigned int u = __float_as_uint(f);
  u += 0x7FFFu + ((u >> 16) & 1u);   // round-to-nearest-even
  return (unsigned short)(u >> 16);
}

static __device__ __forceinline__ v8f wmma_bf16f32(v16bf a, v16bf b, v8f c) {
  // (neg_a, A, neg_b, B, c_mod, C, reuse_a, reuse_b)
  return __builtin_amdgcn_wmma_f32_16x16x32_bf16(false, a, false, b, (short)0, c,
                                                 false, false);
}

// A fragment (16x32 bf16, M x K). src points at (row0, k0) of a row-major
// matrix with leading dimension ld (elements). ISA layout: lane m = lane%16,
// elements 0..7 hold K = half*8 + 0..7, elements 8..15 hold K = 16+half*8+0..7.
static __device__ __forceinline__ v16bf load_frag_A(const unsigned short* src, int ld) {
  int lane = threadIdx.x & 31;
  const unsigned short* p = src + (size_t)(lane & 15) * ld + ((lane >> 4) << 3);
  FragU f;
  f.q[0] = *(const uint4*)p;
  f.q[1] = *(const uint4*)(p + 16);
  return f.v;
}

// B fragment (32x16 bf16, K x N). src points at (n0, k0) of an "n-major"
// matrix: ld elements between consecutive columns n, K contiguous within a
// column. ISA layout: lane n = lane%16, elements 0..7 hold K = half*8 + 0..7, elements 8..15 hold K = 16+half*8+0..7.
static __device__ __forceinline__ v16bf load_frag_B(const unsigned short* src, int ld) {
  int lane = threadIdx.x & 31;
  const unsigned short* p = src + (size_t)(lane & 15) * ld + ((lane >> 4) << 3);
  FragU f;
  f.q[0] = *(const uint4*)p;
  f.q[1] = *(const uint4*)(p + 16);
  return f.v;
}

// ---------------------------------------------------------------------------
// Tensor Data Mover: DMA a 2-D tile (tile_w x tile_h elements of 2B data) from
// a row-major [nrows][row_len] bf16 tensor into LDS, inserting 4 DWORDs of
// LDS padding after every 16 DWORDs (row stride 32 elem -> 40 elem in LDS).
// D# bit layout per CDNA5 ISA ch.8. clang-23 6-arg builtin.
// ---------------------------------------------------------------------------
static __device__ __forceinline__ void tdm_load_tile_2d(
    unsigned lds_byte_addr, const unsigned short* gptr,
    int row_len_elems, int nrows_total, int row_stride_elems,
    int tile_w_elems, int tile_h_rows) {
  unsigned long long ga = (unsigned long long)(uintptr_t)gptr;
  v4u g0;
  g0.x = 1u;                                        // count=1, user descriptor
  g0.y = lds_byte_addr;                             // lds_addr[31:0]
  g0.z = (unsigned)(ga & 0xFFFFFFFFu);              // global_addr[31:0]
  g0.w = (unsigned)((ga >> 32) & 0x1FFFFFFu)        // global_addr[56:32]
         | (2u << 30);                              // type=2 ("image")
  v8i g1;
  // data_size=1 (2B), pad_enable=1, pad_interval=3 (16 DW), pad_amount=3 (4 DW)
  g1[0] = (1 << 16) | (1 << 20) | (3 << 22) | (3 << 25);
  g1[1] = (int)(((unsigned)row_len_elems & 0xFFFFu) << 16);          // dim0 lo16 @ [31:16]
  g1[2] = (int)((((unsigned)row_len_elems >> 16) & 0xFFFFu)          // dim0 hi16
         | (((unsigned)nrows_total & 0xFFFFu) << 16));               // dim1 lo16
  g1[3] = (int)((((unsigned)nrows_total >> 16) & 0xFFFFu)            // dim1 hi16
         | (((unsigned)tile_w_elems & 0xFFFFu) << 16));              // tile_dim0
  g1[4] = (int)((unsigned)tile_h_rows & 0xFFFFu);                    // tile_dim1 (tile_dim2=0)
  g1[5] = row_stride_elems;                                          // dim0_stride lo32
  g1[6] = 0;                                                         // stride hi / dim1_stride
  g1[7] = 0;
  v4i z4;
  z4[0] = 0; z4[1] = 0; z4[2] = 0; z4[3] = 0;
  v8i z8;
  z8[0] = 0; z8[1] = 0; z8[2] = 0; z8[3] = 0;
  z8[4] = 0; z8[5] = 0; z8[6] = 0; z8[7] = 0;
  __builtin_amdgcn_tensor_load_to_lds(g0, g1, z4, z4, z8, 0);
}

// ---------------------------------------------------------------------------
// bf16 x bf16(T) -> f32 GEMM.  C[M,N] = A[M,K] @ Bt[N,K]^T
// BM=BN=128, BK=32; 256 threads (8 waves), wave owns 32x64.  Both tiles are
// TDM-staged (double buffered); wave 0 issues DMA, gates with TENSORcnt.
// ---------------------------------------------------------------------------
#define GBM 128
#define GBN 128
#define GBK 32
#define GLDS 40   // 32 data + 8 pad elements per row (TDM pad: 16DW + 4DW)

__global__ __launch_bounds__(256) void gemm_bf16_wmma_kernel(
    const unsigned short* __restrict__ A,   // [M][K] row-major
    const unsigned short* __restrict__ Bt,  // [N][K] row-major (pre-transposed B)
    float* __restrict__ C,
    int M, int N, int K) {
  __shared__ __align__(16) unsigned short sA[2][GBM * GLDS];
  __shared__ __align__(16) unsigned short sB[2][GBN * GLDS];
  int bm = blockIdx.y * GBM;
  int bn = blockIdx.x * GBN;
  int t = threadIdx.x;
  int wave = t >> 5;
  int wm = (wave & 3) << 5;   // 0,32,64,96
  int wn = (wave >> 2) << 6;  // 0,64
  bool issuer = (t < 32);     // wave 0 drives the TDM (EXEC ignored by TDM)

  v8f acc[2][4];
#pragma unroll
  for (int i = 0; i < 2; ++i)
#pragma unroll
    for (int j = 0; j < 4; ++j)
#pragma unroll
      for (int e = 0; e < 8; ++e) acc[i][j][e] = 0.0f;

  const int nk = K / GBK;

  if (issuer) {
    tdm_load_tile_2d((unsigned)(uintptr_t)&sA[0][0], A + (size_t)bm * K, K, M, K,
                     GBK, GBM);
    tdm_load_tile_2d((unsigned)(uintptr_t)&sB[0][0], Bt + (size_t)bn * K, K, N, K,
                     GBK, GBN);
    __builtin_amdgcn_s_wait_tensorcnt(0);
  }
  __syncthreads();

  for (int ik = 0; ik < nk; ++ik) {
    int cur = ik & 1;
    int nxt = cur ^ 1;
    if (issuer && (ik + 1) < nk) {
      int k0 = (ik + 1) * GBK;
      tdm_load_tile_2d((unsigned)(uintptr_t)&sA[nxt][0],
                       A + (size_t)bm * K + k0, K, M, K, GBK, GBM);
      tdm_load_tile_2d((unsigned)(uintptr_t)&sB[nxt][0],
                       Bt + (size_t)bn * K + k0, K, N, K, GBK, GBN);
    }

    v16bf af[2], bfr[4];
#pragma unroll
    for (int i = 0; i < 2; ++i)
      af[i] = load_frag_A(sA[cur] + (wm + (i << 4)) * GLDS, GLDS);
#pragma unroll
    for (int j = 0; j < 4; ++j)
      bfr[j] = load_frag_B(sB[cur] + (wn + (j << 4)) * GLDS, GLDS);
#pragma unroll
    for (int i = 0; i < 2; ++i)
#pragma unroll
      for (int j = 0; j < 4; ++j)
        acc[i][j] = wmma_bf16f32(af[i], bfr[j], acc[i][j]);

    if (issuer && (ik + 1) < nk) __builtin_amdgcn_s_wait_tensorcnt(0);
    __syncthreads();
  }

  // Epilogue: C layout — lane col = lane%16, VGPR e -> row e + 8*(lane/16)
  int lane = t & 31;
  int col = lane & 15;
  int row16 = (lane >> 4) << 3;
#pragma unroll
  for (int i = 0; i < 2; ++i)
#pragma unroll
    for (int j = 0; j < 4; ++j) {
      float* cp = C + (size_t)(bm + wm + (i << 4)) * N + bn + wn + (j << 4);
#pragma unroll
      for (int e = 0; e < 8; ++e) cp[(size_t)(row16 + e) * N + col] = acc[i][j][e];
    }
}

// ---------------------------------------------------------------------------
// f32 -> bf16 casts (plain, and transposing for weight matrices)
// ---------------------------------------------------------------------------
__global__ void cast_f32_bf16_kernel(const float* __restrict__ src,
                                     unsigned short* __restrict__ dst, int n) {
  int i = blockIdx.x * blockDim.x + threadIdx.x;
  if (i < n) dst[i] = f32_to_bf16_rne(src[i]);
}

// dst[(dstOff + c) * dstLd + r] = src[r * cols + c]  (W[K][N] -> Wt[N][K])
__global__ void cast_transpose_bf16_kernel(const float* __restrict__ src,
                                           unsigned short* __restrict__ dst,
                                           int cols, int dstLd, int dstOff, int n) {
  int i = blockIdx.x * blockDim.x + threadIdx.x;
  if (i < n) {
    int r = i / cols, c = i - r * cols;
    dst[(size_t)(dstOff + c) * dstLd + r] = f32_to_bf16_rne(src[i]);
  }
}

// ---------------------------------------------------------------------------
// RoPE (roll variant) + GQA split/layout + cast to bf16.
// qkv f32 [B*L][3072] (Q:0..2047, K:2048..2559, V:2560..3071)
// Qb [B][16][L][128] (pre-scaled by 1/sqrt(128)), Kb [B][4][L][128],
// Vt [B][4][128][L] (transposed so attention B-frags are contiguous).
// ---------------------------------------------------------------------------
__global__ __launch_bounds__(256) void rope_split_kernel(
    const float* __restrict__ qkv,
    unsigned short* __restrict__ Qb,
    unsigned short* __restrict__ Kb,
    unsigned short* __restrict__ Vt,
    int L) {
  int token = blockIdx.x;
  int b = token / L;
  int l = token - b * L;
  const float* row = qkv + (size_t)token * 3072;
  float pos = (float)l;
  const float qscale = 0.08838834764831845f;  // 1/sqrt(128)
  const float nln = -9.210340371976184f;      // -ln(10000)

  for (int i = threadIdx.x; i < 2048; i += 256) {
    int h = i >> 7, d = i & 127;
    float inv = __expf(nln * (float)((d & 63) << 1) * (1.0f / 128.0f));
    float sn, cs;
    __sincosf(pos * inv, &sn, &cs);
    int dm1 = (d + 127) & 127;  // roll(shift=1): rot[d] = x[d]*cos + x[d-1]*sin
    float r = row[i] * cs + row[(h << 7) + dm1] * sn;
    Qb[(((size_t)(b * 16 + h)) * L + l) * 128 + d] = f32_to_bf16_rne(r * qscale);
  }
  for (int i = threadIdx.x; i < 512; i += 256) {
    int h = i >> 7, d = i & 127;
    float inv = __expf(nln * (float)((d & 63) << 1) * (1.0f / 128.0f));
    float sn, cs;
    __sincosf(pos * inv, &sn, &cs);
    int dm1 = (d + 127) & 127;
    float r = row[2048 + i] * cs + row[2048 + (h << 7) + dm1] * sn;
    Kb[(((size_t)(b * 4 + h)) * L + l) * 128 + d] = f32_to_bf16_rne(r);
  }
  for (int i = threadIdx.x; i < 512; i += 256) {
    int h = i >> 7, d = i & 127;
    Vt[(((size_t)(b * 4 + h)) * 128 + d) * L + l] = f32_to_bf16_rne(row[2560 + i]);
  }
}

// ---------------------------------------------------------------------------
// Flash attention, 128 threads = 4 waves, each wave owns 16 query rows.
// grid = B*16*(L/64). kv head = h/4 (repeat_interleave GQA).
// ---------------------------------------------------------------------------
__global__ __launch_bounds__(128) void gqa_attn_kernel(
    const unsigned short* __restrict__ Qb,
    const unsigned short* __restrict__ Kb,
    const unsigned short* __restrict__ Vt,
    unsigned short* __restrict__ Ob,   // [B*L][2048] bf16
    int L) {
  int nqt = L >> 6;
  int qt = blockIdx.x % nqt;
  int bh = blockIdx.x / nqt;
  int h = bh & 15;
  int b = bh >> 4;
  int kvh = h >> 2;
  int wave = threadIdx.x >> 5;
  int lane = threadIdx.x & 31;
  int q0 = (qt << 6) + (wave << 4);

  const unsigned short* Qp = Qb + ((size_t)(b * 16 + h) * L + q0) * 128;
  const unsigned short* Kp = Kb + (size_t)(b * 4 + kvh) * L * 128;
  const unsigned short* Vp = Vt + (size_t)(b * 4 + kvh) * 128 * L;

  __shared__ __align__(16) unsigned short sP[4][16 * 32];
  unsigned short* myP = sP[wave];

  v16bf qf[4];
#pragma unroll
  for (int c = 0; c < 4; ++c) qf[c] = load_frag_A(Qp + c * 32, 128);

  v8f o[8];
  float mrow[8], lrow[8];
#pragma unroll
  for (int f = 0; f < 8; ++f)
#pragma unroll
    for (int j = 0; j < 8; ++j) o[f][j] = 0.0f;
#pragma unroll
  for (int j = 0; j < 8; ++j) { mrow[j] = -3.0e38f; lrow[j] = 0.0f; }

  int row16 = (lane >> 4) << 3;
  int col = lane & 15;

  for (int kt = 0; kt < L; kt += 32) {
    v8f s0, s1;
#pragma unroll
    for (int j = 0; j < 8; ++j) { s0[j] = 0.0f; s1[j] = 0.0f; }
#pragma unroll
    for (int c = 0; c < 4; ++c) {
      v16bf kb0 = load_frag_B(Kp + (size_t)kt * 128 + c * 32, 128);
      v16bf kb1 = load_frag_B(Kp + (size_t)(kt + 16) * 128 + c * 32, 128);
      s0 = wmma_bf16f32(qf[c], kb0, s0);
      s1 = wmma_bf16f32(qf[c], kb1, s1);
    }
    // online softmax: rows live at VGPR j across 16-lane halves
    float tmax[8];
#pragma unroll
    for (int j = 0; j < 8; ++j) tmax[j] = fmaxf(s0[j], s1[j]);
#pragma unroll
    for (int off = 1; off < 16; off <<= 1)
#pragma unroll
      for (int j = 0; j < 8; ++j)
        tmax[j] = fmaxf(tmax[j], __shfl_xor(tmax[j], off, 32));

    float scl[8], rsum[8];
#pragma unroll
    for (int j = 0; j < 8; ++j) {
      float mn = fmaxf(mrow[j], tmax[j]);
      scl[j] = __expf(mrow[j] - mn);
      mrow[j] = mn;
      float p0 = __expf(s0[j] - mn);
      float p1 = __expf(s1[j] - mn);
      rsum[j] = p0 + p1;
      // bounce P through LDS: C-layout (lane=col) -> A-layout (lane=row)
      myP[(row16 + j) * 32 + col]      = f32_to_bf16_rne(p0);
      myP[(row16 + j) * 32 + col + 16] = f32_to_bf16_rne(p1);
    }
#pragma unroll
    for (int off = 1; off < 16; off <<= 1)
#pragma unroll
      for (int j = 0; j < 8; ++j) rsum[j] += __shfl_xor(rsum[j], off, 32);
#pragma unroll
    for (int j = 0; j < 8; ++j) lrow[j] = lrow[j] * scl[j] + rsum[j];
#pragma unroll
    for (int f = 0; f < 8; ++f)
#pragma unroll
      for (int j = 0; j < 8; ++j) o[f][j] *= scl[j];

    asm volatile("s_wait_dscnt 0" ::: "memory");  // wave-local LDS RAW fence
    v16bf pf = load_frag_A(myP, 32);
#pragma unroll
    for (int f = 0; f < 8; ++f) {
      v16bf vb = load_frag_B(Vp + (size_t)(f * 16) * L + kt, L);
      o[f] = wmma_bf16f32(pf, vb, o[f]);
    }
  }

  float invl[8];
#pragma unroll
  for (int j = 0; j < 8; ++j) invl[j] = 1.0f / lrow[j];
#pragma unroll
  for (int f = 0; f < 8; ++f)
#pragma unroll
    for (int j = 0; j < 8; ++j) {
      int q = q0 + row16 + j;
      int c2 = h * 128 + f * 16 + col;
      Ob[(size_t)(b * L + q) * 2048 + c2] = f32_to_bf16_rne(o[f][j] * invl[j]);
    }
}

// ---------------------------------------------------------------------------
// Host orchestration
// ---------------------------------------------------------------------------
extern "C" void kernel_launch(void* const* d_in, const int* in_sizes, int n_in,
                              void* d_out, int out_size, void* d_ws, size_t ws_size,
                              hipStream_t stream) {
  (void)in_sizes; (void)n_in; (void)out_size; (void)ws_size;
  const float* x  = (const float*)d_in[0];
  const float* Wq = (const float*)d_in[1];
  const float* Wk = (const float*)d_in[2];
  const float* Wv = (const float*)d_in[3];
  const float* Wo = (const float*)d_in[4];

  const int B = 2, L = 2048, D = 2048;
  const int M = B * L;        // 4096
  const int NQKV = 3072;      // 2048 Q + 512 K + 512 V

  char* ws = (char*)d_ws;
  size_t off = 0;
  auto alloc = [&](size_t bytes) {
    char* p = ws + off;
    off += (bytes + 255) & ~(size_t)255;
    return p;
  };

  unsigned short* xb     = (unsigned short*)alloc((size_t)M * D * 2);     // 16 MB
  unsigned short* WqkvT  = (unsigned short*)alloc((size_t)NQKV * D * 2);  // 12 MB  [3072][2048]
  unsigned short* WoT    = (unsigned short*)alloc((size_t)D * D * 2);     //  8 MB  [2048][2048]
  float*          qkv    = (float*)alloc((size_t)M * NQKV * 4);           // 48 MB
  unsigned short* Qb     = (unsigned short*)alloc((size_t)B * 16 * L * 128 * 2);
  unsigned short* Kb     = (unsigned short*)alloc((size_t)B * 4 * L * 128 * 2);
  unsigned short* Vt     = (unsigned short*)alloc((size_t)B * 4 * 128 * L * 2);
  unsigned short* Ab     = (unsigned short*)alloc((size_t)M * D * 2);     // 16 MB

  int n;
  n = M * D;
  cast_f32_bf16_kernel<<<(n + 255) / 256, 256, 0, stream>>>(x, xb, n);
  // Weights cast + transposed so GEMM B operand is [N][K] (TDM-friendly)
  n = D * 2048;
  cast_transpose_bf16_kernel<<<(n + 255) / 256, 256, 0, stream>>>(Wq, WqkvT, 2048, D, 0, n);
  n = D * 512;
  cast_transpose_bf16_kernel<<<(n + 255) / 256, 256, 0, stream>>>(Wk, WqkvT, 512, D, 2048, n);
  cast_transpose_bf16_kernel<<<(n + 255) / 256, 256, 0, stream>>>(Wv, WqkvT, 512, D, 2560, n);
  n = D * D;
  cast_transpose_bf16_kernel<<<(n + 255) / 256, 256, 0, stream>>>(Wo, WoT, D, D, 0, n);

  // QKV = x @ [Wq|Wk|Wv]
  dim3 g1(NQKV / GBN, M / GBM);
  gemm_bf16_wmma_kernel<<<g1, 256, 0, stream>>>(xb, WqkvT, qkv, M, NQKV, D);

  // RoPE + split (Q pre-scaled by 1/sqrt(HD)), V transposed
  rope_split_kernel<<<M, 256, 0, stream>>>(qkv, Qb, Kb, Vt, L);

  // Flash attention -> Ab bf16 [B*L][2048]
  gqa_attn_kernel<<<B * 16 * (L / 64), 128, 0, stream>>>(Qb, Kb, Vt, Ab, L);

  // out = Ab @ Wo (f32 result straight to d_out)
  dim3 g2(D / GBN, M / GBM);
  gemm_bf16_wmma_kernel<<<g2, 256, 0, stream>>>(Ab, WoT, (float*)d_out, M, D, D);
}
